// MGC_Combiner_41918880809190
// MI455X (gfx1250) — matrix operand order repaired and checked by reading, not compiled, not run on hardware
//
#include <hip/hip_runtime.h>
#include <hip/hip_bf16.h>

typedef __attribute__((ext_vector_type(16))) __bf16 v16bf;
typedef __attribute__((ext_vector_type(8)))  float  v8f;

#define LSTR 136   // bf16 elements per LDS row: 128 + 8 pad -> 272B stride (conflict-free b128 frags)

// ---------------------------------------------------------------------------
// Prep: convert W2/W4 (128x128 f32 row-major) to bf16 in workspace
// ---------------------------------------------------------------------------
__global__ __launch_bounds__(256) void cvt_weights_kernel(
    const float* __restrict__ w2f, const float* __restrict__ w4f,
    __bf16* __restrict__ w2b, __bf16* __restrict__ w4b)
{
    int idx = blockIdx.x * 256 + threadIdx.x;          // 32 blocks -> 8192 threads
    for (int k = idx; k < 128 * 128; k += 8192) {
        w2b[k] = (__bf16)w2f[k];
        w4b[k] = (__bf16)w4f[k];
    }
}

// ---------------------------------------------------------------------------
// Prep: proj_i = old_nodes @ W3[:, :128]^T + b3 ; proj_j = old_nodes @ W3[:,128:]^T
// One block per (b,n) row.  W3_w is (128, 256) row-major.
// ---------------------------------------------------------------------------
__global__ __launch_bounds__(256) void proj_kernel(
    const float* __restrict__ old_nodes, const float* __restrict__ w3w,
    const float* __restrict__ w3b,
    float* __restrict__ proj_i, float* __restrict__ proj_j)
{
    __shared__ float sx[128];
    const int row = blockIdx.x;            // b*256 + n
    const int t = threadIdx.x;
    if (t < 128) sx[t] = old_nodes[(size_t)row * 128 + t];
    __syncthreads();
    if (t < 128) {
        const float* w = w3w + (size_t)t * 256;         // W3_i row t
        float acc = w3b[t];                             // fold b3 into proj_i
        #pragma unroll 4
        for (int k = 0; k < 128; ++k) acc += sx[k] * w[k];
        proj_i[(size_t)row * 128 + t] = acc;
    } else {
        const int o = t - 128;
        const float* w = w3w + (size_t)o * 256 + 128;   // W3_j row o
        float acc = 0.0f;
        #pragma unroll 4
        for (int k = 0; k < 128; ++k) acc += sx[k] * w[k];
        proj_j[(size_t)row * 128 + o] = acc;
    }
}

// ---------------------------------------------------------------------------
// Node path: act_nodes = relu(W1 @ [relu(W0 x + b0); msg] + b1). One block/row.
// ---------------------------------------------------------------------------
__global__ __launch_bounds__(256) void nodes_kernel(
    const float* __restrict__ old_nodes, const float* __restrict__ msgs,
    const float* __restrict__ w0w, const float* __restrict__ w0b,
    const float* __restrict__ w1w, const float* __restrict__ w1b,
    float* __restrict__ out)
{
    __shared__ float sx[128], sm[128], sh0[128], spart[128];
    const int row = blockIdx.x;
    const int t = threadIdx.x;
    if (t < 128) sx[t] = old_nodes[(size_t)row * 128 + t];
    else         sm[t - 128] = msgs[(size_t)row * 128 + (t - 128)];
    __syncthreads();
    if (t < 128) {
        const float* w = w0w + (size_t)t * 128;
        float acc = w0b[t];
        #pragma unroll 4
        for (int k = 0; k < 128; ++k) acc += sx[k] * w[k];
        sh0[t] = fmaxf(acc, 0.0f);
    } else {
        const int o = t - 128;
        const float* w = w1w + (size_t)o * 256 + 128;   // messages half of W1
        float acc = 0.0f;
        #pragma unroll 4
        for (int k = 0; k < 128; ++k) acc += sm[k] * w[k];
        spart[o] = acc;
    }
    __syncthreads();
    if (t < 128) {
        const float* w = w1w + (size_t)t * 256;         // h0 half of W1
        float acc = w1b[t] + spart[t];
        #pragma unroll 4
        for (int k = 0; k < 128; ++k) acc += sh0[k] * w[k];
        out[(size_t)row * 128 + t] = fmaxf(acc, 0.0f);
    }
}

// ---------------------------------------------------------------------------
// Edge path (dominant): per WG, 128 edge rows.
//   h2   = relu(X @ W2^T + b2)            (WMMA bf16, f32 accum)
//   a2   = h2 + relu(proj_i + proj_j)     (b3 already folded into proj_i)
//   out  = relu(a2 @ W4^T + b4)           (WMMA bf16, f32 accum)
// ---------------------------------------------------------------------------
__global__ __launch_bounds__(256) void edges_kernel(
    const float*  __restrict__ old_edges,   // (8, 65536, 128)
    const float*  __restrict__ proj_i,      // (8, 256, 128), includes b3
    const float*  __restrict__ proj_j,      // (8, 256, 128)
    const __bf16* __restrict__ w2b,         // (128,128) row-major [o][k]
    const __bf16* __restrict__ w4b,
    const float*  __restrict__ b2,
    const float*  __restrict__ b4,
    float* __restrict__ out)                // (8, 65536, 128)
{
    __shared__ __bf16 sW2[128 * LSTR];
    __shared__ __bf16 sW4[128 * LSTR];
    __shared__ __bf16 sX [128 * LSTR];      // A tile; reused in place as A2 for GEMM2

    const int tid = threadIdx.x;
    const int blk = blockIdx.x;
    const int b   = blk >> 9;               // / 512 tiles per batch
    const int t   = blk & 511;
    const int e0  = t << 7;                 // first edge row of tile
    const int i   = e0 >> 8;                // source node index (fixed per tile)
    const int j0  = e0 & 255;               // first dest node index

    // ---- stage weights (bf16 -> LDS, padded stride) ----
    {
        const int row  = tid >> 1;
        const int half = (tid & 1) * 64;    // in bf16 elements (=128B)
        const uint4* s2 = (const uint4*)(w2b + row * 128 + half);
        const uint4* s4 = (const uint4*)(w4b + row * 128 + half);
        uint4* d2 = (uint4*)(sW2 + row * LSTR + half);
        uint4* d4 = (uint4*)(sW4 + row * LSTR + half);
        #pragma unroll
        for (int c = 0; c < 8; ++c) { d2[c] = s2[c]; d4[c] = s4[c]; }
    }
    // ---- stage X tile: old_edges f32 -> bf16 LDS ----
    {
        const int row  = tid >> 1;
        const int half = (tid & 1) * 64;
        const float4* src =
            (const float4*)(old_edges + ((size_t)b * 65536 + e0 + row) * 128 + half);
        #pragma unroll
        for (int c = 0; c < 8; ++c) {
            float4 f0 = src[2 * c];
            float4 f1 = src[2 * c + 1];
            union { __bf16 h[8]; uint4 q; } pk;
            pk.h[0] = (__bf16)f0.x; pk.h[1] = (__bf16)f0.y;
            pk.h[2] = (__bf16)f0.z; pk.h[3] = (__bf16)f0.w;
            pk.h[4] = (__bf16)f1.x; pk.h[5] = (__bf16)f1.y;
            pk.h[6] = (__bf16)f1.z; pk.h[7] = (__bf16)f1.w;
            *(uint4*)(sX + row * LSTR + half + c * 8) = pk.q;
        }
    }
    __syncthreads();

    const int wave  = tid >> 5;
    const int lane  = tid & 31;
    const int lhalf = lane >> 4;            // which half-wave
    const int lmod  = lane & 15;
    const int rbase = wave * 16;            // wave-private 16 rows

    // ================= GEMM1: h2_pre = X @ W2^T =================
    v8f acc[8] = {};
    #pragma unroll
    for (int ks = 0; ks < 4; ++ks) {
        // A frag: row (rbase+lmod), K chunks [k0..k0+7] and [k0+16..k0+23], k0 = ks*32 + 8*lhalf
        union { uint4 q[2]; v16bf v; } a;
        const __bf16* ap = sX + (rbase + lmod) * LSTR + ks * 32 + lhalf * 8;
        a.q[0] = *(const uint4*)(ap);
        a.q[1] = *(const uint4*)(ap + 16);
        #pragma unroll
        for (int nt = 0; nt < 8; ++nt) {
            // B frag: W row n = nt*16+lmod, K chunks [k0..k0+7] and [k0+16..k0+23], k0 = ks*32 + 8*lhalf
            union { uint4 q[2]; v16bf v; } bb;
            const __bf16* bp = sW2 + (nt * 16 + lmod) * LSTR + ks * 32 + lhalf * 8;
            bb.q[0] = *(const uint4*)(bp);
            bb.q[1] = *(const uint4*)(bp + 16);
            acc[nt] = __builtin_amdgcn_wmma_f32_16x16x32_bf16(
                false, a.v, false, bb.v, (short)0, acc[nt], false, false);
        }
    }

    // ---- epilogue 1: +b2, relu, + relu(proj_i + proj_j), bf16 back into sX ----
    const float* pi = proj_i + ((size_t)b * 256 + i) * 128;
    const float* pj = proj_j + (size_t)b * 256 * 128;
    #pragma unroll
    for (int nt = 0; nt < 8; ++nt) {
        const int col = nt * 16 + lmod;
        const float bias2 = b2[col];
        const float piv   = pi[col];
        #pragma unroll
        for (int r = 0; r < 8; ++r) {
            const int rl = rbase + r + lhalf * 8;          // local row == dest-node offset
            float h2 = fmaxf(acc[nt][r] + bias2, 0.0f);
            float h3 = fmaxf(piv + pj[(size_t)(j0 + rl) * 128 + col], 0.0f);
            sX[rl * LSTR + col] = (__bf16)(h2 + h3);       // wave-private rows: no barrier
        }
    }

    // ================= GEMM2: out_pre = A2 @ W4^T =================
    v8f acc2[8] = {};
    #pragma unroll
    for (int ks = 0; ks < 4; ++ks) {
        union { uint4 q[2]; v16bf v; } a;
        const __bf16* ap = sX + (rbase + lmod) * LSTR + ks * 32 + lhalf * 8;
        a.q[0] = *(const uint4*)(ap);
        a.q[1] = *(const uint4*)(ap + 16);
        #pragma unroll
        for (int nt = 0; nt < 8; ++nt) {
            union { uint4 q[2]; v16bf v; } bb;
            const __bf16* bp = sW4 + (nt * 16 + lmod) * LSTR + ks * 32 + lhalf * 8;
            bb.q[0] = *(const uint4*)(bp);
            bb.q[1] = *(const uint4*)(bp + 16);
            acc2[nt] = __builtin_amdgcn_wmma_f32_16x16x32_bf16(
                false, a.v, false, bb.v, (short)0, acc2[nt], false, false);
        }
    }

    // ---- epilogue 2: +b4, relu, store fp32 ----
    float* op = out + ((size_t)b * 65536 + e0) * 128;
    #pragma unroll
    for (int nt = 0; nt < 8; ++nt) {
        const int col = nt * 16 + lmod;
        const float bias4 = b4[col];
        #pragma unroll
        for (int r = 0; r < 8; ++r) {
            const int rl = rbase + r + lhalf * 8;
            op[(size_t)rl * 128 + col] = fmaxf(acc2[nt][r] + bias4, 0.0f);
        }
    }
}

// ---------------------------------------------------------------------------
extern "C" void kernel_launch(void* const* d_in, const int* in_sizes, int n_in,
                              void* d_out, int out_size, void* d_ws, size_t ws_size,
                              hipStream_t stream) {
    const float* old_nodes = (const float*)d_in[0];
    const float* old_edges = (const float*)d_in[1];
    const float* messages  = (const float*)d_in[2];
    const float* W0_w = (const float*)d_in[3];
    const float* W0_b = (const float*)d_in[4];
    const float* W1_w = (const float*)d_in[5];
    const float* W1_b = (const float*)d_in[6];
    const float* W2_w = (const float*)d_in[7];
    const float* W2_b = (const float*)d_in[8];
    const float* W3_w = (const float*)d_in[9];
    const float* W3_b = (const float*)d_in[10];
    const float* W4_w = (const float*)d_in[11];
    const float* W4_b = (const float*)d_in[12];

    float* out        = (float*)d_out;
    float* act_nodes  = out;                       // (8,256,128)
    float* act_edges  = out + (size_t)8 * 256 * 128; // (8,65536,128)

    // workspace layout
    float*  proj_i = (float*)d_ws;                 // 262144 f32
    float*  proj_j = proj_i + 262144;              // 262144 f32
    __bf16* w2b    = (__bf16*)(proj_j + 262144);   // 16384 bf16
    __bf16* w4b    = w2b + 16384;                  // 16384 bf16

    cvt_weights_kernel<<<32, 256, 0, stream>>>(W2_w, W4_w, w2b, w4b);
    proj_kernel<<<8 * 256, 256, 0, stream>>>(old_nodes, W3_w, W3_b, proj_i, proj_j);
    nodes_kernel<<<8 * 256, 256, 0, stream>>>(old_nodes, messages, W0_w, W0_b,
                                              W1_w, W1_b, act_nodes);
    edges_kernel<<<8 * 512, 256, 0, stream>>>(old_edges, proj_i, proj_j, w2b, w4b,
                                              W2_b, W4_b, act_edges);
}
